// SelectiveAttention_88235808129251
// MI455X (gfx1250) — compile-verified, matrix operand order checked by reading, not run on hardware
//
#include <hip/hip_runtime.h>

// ---------------------------------------------------------------------------
// NSA-style selective attention for gfx1250 (MI455X), wave32 + WMMA bf16.
//   b=1, m=n=2048, d=128, NHEADS=32, NKV=8 (group g=4),
//   SELECT_SIZE=64 -> S=32 blocks, TOP_N=16, KERNEL=32, STRIDE=16 -> C=127.
//
// Pipeline:
//   (0) nsa_convert_kernel : K -> bf16 [kv][n][d], V -> bf16 transposed
//       [kv][d][n] in workspace (one pass, ~25MB traffic, ~1us at 23.3TB/s).
//   (1) nsa_select_kernel  : sparse-W block scoring + causal top-16 bitmask.
//   (2) nsa_attn_kernel    : flash attention over selected 64-key blocks.
//       Workgroup = 256 threads = 8 waves = 4 heads x 2 row-tiles sharing one
//       K/VT LDS stage (halves L2 traffic vs 16-row workgroups).
//       K/V staged with global_load_async_to_lds_b128 (ASYNCcnt), QK^T and PV
//       on v_wmma_f32_16x16x32_bf16, row reductions on v_permlane16_b32.
//
// Workspace: 64KB masks + 4MB Kbf16 + 4MB VTbf16  (requires ws_size >= 8.5MB)
// ---------------------------------------------------------------------------

typedef __bf16 bf16_t;
typedef bf16_t v16bf __attribute__((ext_vector_type(16)));
typedef bf16_t v8bf  __attribute__((ext_vector_type(8)));
typedef float  v8f   __attribute__((ext_vector_type(8)));

#define M_TOT     2048
#define N_TOT     2048
#define D_DIM     128
#define NHEADS    32
#define NKV       8
#define GRP       4
#define SBLK      64      // SELECT_SIZE
#define NSB       32      // S = n / SELECT_SIZE
#define TOPN      16
#define C_CMP     127
#define KEEPV     999999.0f
#define NEGBIG    (-1.0e30f)
// 1/sqrt(128) * log2(e)  (exp2-domain softmax), folded into Q fragments
#define QK_SCALE  (0.08838834764831845f * 1.4426950408889634f)

static __device__ __forceinline__ unsigned int pack2(float a, float b) {
  union { bf16_t h[2]; unsigned int u; } c;
  c.h[0] = (bf16_t)a; c.h[1] = (bf16_t)b;
  return c.u;
}
static __device__ __forceinline__ unsigned short f2bf_u(float f) {
  union { bf16_t h; unsigned short u; } c;
  c.h = (bf16_t)f;
  return c.u;
}

static __device__ __forceinline__ v16bf load_frag16(const unsigned short* p0,
                                                    const unsigned short* p1) {
  v8bf lo = *reinterpret_cast<const v8bf*>(p0);   // ds_load_b128
  v8bf hi = *reinterpret_cast<const v8bf*>(p1);   // ds_load_b128
  return __builtin_shufflevector(lo, hi, 0,1,2,3,4,5,6,7,8,9,10,11,12,13,14,15);
}

// ---- cross-lane reductions within 16-lane halves (VALU, not DS) -----------
#if defined(__gfx1250__) && __has_builtin(__builtin_amdgcn_permlane16)
#define PERML(x, lo, hi) __builtin_amdgcn_permlane16((x), (x), (lo), (hi), false, false)
static __device__ __forceinline__ float rowmax16(float x) {
  union { float f; unsigned u; } a, b; a.f = x;
  b.u = PERML(a.u, 0x67452301u, 0xEFCDAB89u); a.f = fmaxf(a.f, b.f);  // xor 1
  b.u = PERML(a.u, 0x45670123u, 0xCDEF89ABu); a.f = fmaxf(a.f, b.f);  // xor 2
  b.u = PERML(a.u, 0x01234567u, 0x89ABCDEFu); a.f = fmaxf(a.f, b.f);  // xor 4
  b.u = PERML(a.u, 0xFEDCBA98u, 0x76543210u); a.f = fmaxf(a.f, b.f);  // xor 8
  return a.f;
}
static __device__ __forceinline__ float rowsum16(float x) {
  union { float f; unsigned u; } a, b; a.f = x;
  b.u = PERML(a.u, 0x67452301u, 0xEFCDAB89u); a.f += b.f;
  b.u = PERML(a.u, 0x45670123u, 0xCDEF89ABu); a.f += b.f;
  b.u = PERML(a.u, 0x01234567u, 0x89ABCDEFu); a.f += b.f;
  b.u = PERML(a.u, 0xFEDCBA98u, 0x76543210u); a.f += b.f;
  return a.f;
}
#else
static __device__ __forceinline__ float rowmax16(float x) {
  for (int off = 1; off < 16; off <<= 1) x = fmaxf(x, __shfl_xor(x, off, 32));
  return x;
}
static __device__ __forceinline__ float rowsum16(float x) {
  for (int off = 1; off < 16; off <<= 1) x += __shfl_xor(x, off, 32);
  return x;
}
#endif

// ===========================================================================
// Kernel 0: K -> bf16 [kv][n][d]; V -> bf16 transposed [kv][d][n].
// ===========================================================================
__global__ __launch_bounds__(256) void nsa_convert_kernel(
    const float* __restrict__ k, const float* __restrict__ v,
    unsigned short* __restrict__ kbf, unsigned short* __restrict__ vt) {
  __shared__ __align__(16) unsigned short Vsm[64][136];
  const int nb = blockIdx.x, kv = blockIdx.y, t = threadIdx.x;
  {
    int r  = t >> 2;               // n within block: 0..63
    int dq = (t & 3) * 32;         // d quarter
    const float4* kp = reinterpret_cast<const float4*>(
        k + ((size_t)(nb * 64 + r) * NKV + kv) * D_DIM + dq);
    const float4* vp = reinterpret_cast<const float4*>(
        v + ((size_t)(nb * 64 + r) * NKV + kv) * D_DIM + dq);
    uint4* kd = reinterpret_cast<uint4*>(
        kbf + ((size_t)kv * N_TOT + nb * 64 + r) * D_DIM + dq);
#pragma unroll
    for (int i = 0; i < 4; ++i) {
      float4 a = kp[2 * i], b = kp[2 * i + 1];
      uint4 u;
      u.x = pack2(a.x, a.y); u.y = pack2(a.z, a.w);
      u.z = pack2(b.x, b.y); u.w = pack2(b.z, b.w);
      kd[i] = u;
      float4 c = vp[2 * i], d2 = vp[2 * i + 1];
      uint4 w2;
      w2.x = pack2(c.x, c.y); w2.y = pack2(c.z, c.w);
      w2.z = pack2(d2.x, d2.y); w2.w = pack2(d2.z, d2.w);
      *reinterpret_cast<uint4*>(&Vsm[r][dq + 8 * i]) = w2;
    }
  }
  __syncthreads();
  {
    int d  = t >> 1;               // 0..127
    int n0 = (t & 1) * 32;
    unsigned short* dst = vt + ((size_t)kv * D_DIM + d) * N_TOT + nb * 64 + n0;
#pragma unroll
    for (int i = 0; i < 4; ++i) {
      uint4 u;
      u.x = (unsigned)Vsm[n0 + 8 * i + 0][d] | ((unsigned)Vsm[n0 + 8 * i + 1][d] << 16);
      u.y = (unsigned)Vsm[n0 + 8 * i + 2][d] | ((unsigned)Vsm[n0 + 8 * i + 3][d] << 16);
      u.z = (unsigned)Vsm[n0 + 8 * i + 4][d] | ((unsigned)Vsm[n0 + 8 * i + 5][d] << 16);
      u.w = (unsigned)Vsm[n0 + 8 * i + 6][d] | ((unsigned)Vsm[n0 + 8 * i + 7][d] << 16);
      *reinterpret_cast<uint4*>(dst + 8 * i) = u;
    }
  }
}

// ===========================================================================
// Kernel 1: compressed-block -> select-block scoring + causal top-k -> bitmask
// ===========================================================================
__global__ __launch_bounds__(256) void nsa_select_kernel(
    const float* __restrict__ p, unsigned int* __restrict__ msk) {
  int t = blockIdx.x * blockDim.x + threadIdx.x;
  if (t >= NKV * M_TOT) return;
  int kv  = t / M_TOT;
  int row = t - kv * M_TOT;

  float sp[NSB];
#pragma unroll
  for (int s = 0; s < NSB; ++s) sp[s] = -3.0e38f;

  for (int gi = 0; gi < GRP; ++gi) {
    const float* pr = p + ((size_t)(kv * GRP + gi) * M_TOT + row) * C_CMP;
#pragma unroll
    for (int s = 0; s < NSB; ++s) {
      int c0 = 4 * s;
      float v = 2.0f * (pr[c0] + pr[c0 + 1] + pr[c0 + 2]);
      if (c0 - 1 >= 0)     v += pr[c0 - 1];
      if (c0 + 3 < C_CMP)  v += pr[c0 + 3];
      sp[s] = fmaxf(sp[s], v);
    }
  }

  int rb = row >> 6;
  sp[0] = KEEPV;
  sp[rb] = KEEPV;
  sp[(rb - 1) > 0 ? (rb - 1) : 0] = KEEPV;

  // stable top-k (lowest index wins ties), keep first min(TOPN, rb+1) ranks
  int nk = (rb + 1) < TOPN ? (rb + 1) : TOPN;
  unsigned int used = 0u, out = 0u;
  for (int r = 0; r < nk; ++r) {
    float best = -3.4e38f; int bi = 0;
#pragma unroll
    for (int s = 0; s < NSB; ++s) {
      bool cand = !((used >> s) & 1u) && (sp[s] > best);
      best = cand ? sp[s] : best;
      bi   = cand ? s     : bi;
    }
    used |= (1u << bi);
    out  |= (1u << bi);
  }
  msk[kv * M_TOT + row] = out;
}

// ===========================================================================
// Kernel 2: flash attention over selected 64-key blocks with WMMA bf16.
//   grid = (m/32, NKV); 256 threads = 8 waves:
//     wave w -> head (kv*4 + (w&3)), row-tile rt=(w>>2) of the 32-row range.
//   K staged [64][128] bf16 (row pad 136), VT staged [128][64] bf16 (pad 72).
// ===========================================================================
#define KSTRIDE 136   // shorts per K row  (272 B rows, 16B-aligned, bank-spread)
#define VSTRIDE 72    // shorts per VT row (144 B)
#define PSTRIDE 72    // shorts per P row  (144 B)

struct SMem {
  unsigned short Ksm[SBLK][KSTRIDE];      // 17408 B
  unsigned short VTsm[D_DIM][VSTRIDE];    // 18432 B
  unsigned short Psm[8][16][PSTRIDE];     // 18432 B (per-wave P tiles)
  unsigned int   rmask[32];
};

__global__ __launch_bounds__(256) void nsa_attn_kernel(
    const float* __restrict__ q, const unsigned short* __restrict__ kbf,
    const unsigned short* __restrict__ vt, const unsigned int* __restrict__ msk,
    float* __restrict__ out) {
  __shared__ __align__(16) SMem sm;

  const int wgrow0 = blockIdx.x * 32;
  const int kv   = blockIdx.y;
  const int tid  = threadIdx.x;
  const int w    = tid >> 5;            // wave id
  const int rt   = w >> 2;              // row-tile 0/1 within the 32-row range
  const int lane = tid & 31;
  const int l15  = lane & 15;
  const int half8 = (lane & 16) >> 1;   // 0 or 8  (C/D-layout row offset)
  const int klo   = (lane & 16) >> 1;   // 0 or 8  (A/B-layout K offset)
  const int h    = kv * GRP + (w & 3);
  const int row0 = wgrow0 + rt * 16;

  if (tid < 32) sm.rmask[tid] = msk[kv * M_TOT + wgrow0 + tid];

  // ---- Q A-fragments (16 rows x 128, 4 K-chunks of 32), scale folded in ---
  const float* qbase = q + ((size_t)(row0 + l15) * NHEADS + h) * D_DIM;
  v16bf qf[4];
#pragma unroll
  for (int ck = 0; ck < 4; ++ck) {
    int d0 = ck * 32 + klo;
    const float4* a = reinterpret_cast<const float4*>(qbase + d0);
    const float4* b = reinterpret_cast<const float4*>(qbase + d0 + 16);
    float4 f0 = a[0], f1 = a[1], f2 = b[0], f3 = b[1];
    v16bf t;
    t[0]=(bf16_t)(f0.x*QK_SCALE); t[1]=(bf16_t)(f0.y*QK_SCALE);
    t[2]=(bf16_t)(f0.z*QK_SCALE); t[3]=(bf16_t)(f0.w*QK_SCALE);
    t[4]=(bf16_t)(f1.x*QK_SCALE); t[5]=(bf16_t)(f1.y*QK_SCALE);
    t[6]=(bf16_t)(f1.z*QK_SCALE); t[7]=(bf16_t)(f1.w*QK_SCALE);
    t[8]=(bf16_t)(f2.x*QK_SCALE); t[9]=(bf16_t)(f2.y*QK_SCALE);
    t[10]=(bf16_t)(f2.z*QK_SCALE); t[11]=(bf16_t)(f2.w*QK_SCALE);
    t[12]=(bf16_t)(f3.x*QK_SCALE); t[13]=(bf16_t)(f3.y*QK_SCALE);
    t[14]=(bf16_t)(f3.z*QK_SCALE); t[15]=(bf16_t)(f3.w*QK_SCALE);
    qf[ck] = t;
  }

  __syncthreads();

  // per-row-tile block unions + causal bounds (computed identically by all)
  unsigned int um0 = 0u, um1 = 0u;
#pragma unroll
  for (int i = 0; i < 16; ++i) { um0 |= sm.rmask[i]; um1 |= sm.rmask[16 + i]; }
  const int sMax0 = (wgrow0 + 15) >> 6;
  const int sMax1 = (wgrow0 + 31) >> 6;
  const unsigned int umW = rt ? um1 : um0;
  const int sMaxW = rt ? sMax1 : sMax0;

  // per-lane row masks in registers (asm memory clobbers would force LDS reloads)
  unsigned int mreg[8];
#pragma unroll
  for (int j = 0; j < 8; ++j) mreg[j] = sm.rmask[rt * 16 + j + half8];

  v8f oacc[8];
#pragma unroll
  for (int dt = 0; dt < 8; ++dt) oacc[dt] = {};
  float mrun[8], lrun[8];
#pragma unroll
  for (int j = 0; j < 8; ++j) { mrun[j] = NEGBIG; lrun[j] = 0.0f; }

  for (int s = 0; s <= sMax1; ++s) {
    bool need0 = (s <= sMax0) && ((um0 >> s) & 1u);
    bool need1 = ((um1 >> s) & 1u);
    if (!(need0 || need1)) continue;          // uniform across workgroup
    __syncthreads();   // previous block's LDS reads done

    // ---- stage K (64x128) and VT (128x64) bf16 blocks into LDS -----------
#if defined(__gfx1250__)
    {
      // async DMA, 16B per lane per op, ASYNCcnt-tracked (no VGPR roundtrip)
      unsigned long long kgb = (unsigned long long)(const void*)(
          kbf + ((size_t)kv * N_TOT + s * SBLK) * D_DIM);
      unsigned int lbK = (unsigned int)(size_t)(&sm.Ksm[0][0]);
      int r = tid >> 2, qd = (tid & 3) * 64;
#pragma unroll
      for (int i = 0; i < 4; ++i) {
        unsigned int go = (unsigned)(r * 256 + qd + i * 16);
        unsigned int la = lbK + (unsigned)(r * 272 + qd + i * 16);
        asm volatile("global_load_async_to_lds_b128 %0, %1, %2"
                     :: "v"(la), "v"(go), "s"(kgb) : "memory");
      }
      unsigned long long vgb = (unsigned long long)(const void*)(
          vt + (size_t)kv * D_DIM * N_TOT + s * SBLK);
      unsigned int lbV = (unsigned int)(size_t)(&sm.VTsm[0][0]);
      int d = tid >> 1, hn = (tid & 1) * 64;
#pragma unroll
      for (int i = 0; i < 4; ++i) {
        unsigned int go = (unsigned)(d * 4096 + hn + i * 16);
        unsigned int la = lbV + (unsigned)(d * 144 + hn + i * 16);
        asm volatile("global_load_async_to_lds_b128 %0, %1, %2"
                     :: "v"(la), "v"(go), "s"(vgb) : "memory");
      }
      asm volatile("s_wait_asynccnt 0x0" ::: "memory");
    }
#else
    {
      int r = tid >> 2, qd = (tid & 3) * 32;   // qd in shorts
      const uint4* gk = reinterpret_cast<const uint4*>(
          kbf + ((size_t)kv * N_TOT + s * SBLK + r) * D_DIM + qd);
      uint4* lk = reinterpret_cast<uint4*>(&sm.Ksm[r][qd]);
      for (int i = 0; i < 4; ++i) lk[i] = gk[i];
      int d = tid >> 1, hn = (tid & 1) * 32;   // hn in shorts
      const uint4* gv = reinterpret_cast<const uint4*>(
          vt + ((size_t)kv * D_DIM + d) * N_TOT + s * SBLK + hn);
      uint4* lv = reinterpret_cast<uint4*>(&sm.VTsm[d][hn]);
      for (int i = 0; i < 4; ++i) lv[i] = gv[i];
    }
#endif
    __syncthreads();

    if (!(rt ? need1 : need0)) continue;      // per-wave skip (no barriers below)

    // ---- S = Q K^T : 4 N-tiles x 4 K-chunks of WMMA bf16 -----------------
    v8f cf[4];
#pragma unroll
    for (int nt = 0; nt < 4; ++nt) {
      v8f acc = {};
      const unsigned short* krow = &sm.Ksm[nt * 16 + l15][0];
#pragma unroll
      for (int ck = 0; ck < 4; ++ck) {
        const unsigned short* p0 = krow + ck * 32 + klo;
        v16bf bfrag = load_frag16(p0, p0 + 16);
        acc = __builtin_amdgcn_wmma_f32_16x16x32_bf16(
            false, qf[ck], false, bfrag, (short)0, acc, false, false);
      }
      cf[nt] = acc;
    }

    // ---- mask (causal + per-row block select) ----------------------------
#pragma unroll
    for (int nt = 0; nt < 4; ++nt) {
      int key = s * SBLK + nt * 16 + l15;
#pragma unroll
      for (int j = 0; j < 8; ++j) {
        bool keep = (key <= row0 + j + half8) && ((mreg[j] >> s) & 1u);
        cf[nt][j] = keep ? cf[nt][j] : NEGBIG;
      }
    }

    // ---- online softmax (row reductions across 16-lane halves) -----------
    float mnew[8], alpha[8];
#pragma unroll
    for (int j = 0; j < 8; ++j) {
      float mr = rowmax16(fmaxf(fmaxf(cf[0][j], cf[1][j]),
                                fmaxf(cf[2][j], cf[3][j])));
      mnew[j]  = fmaxf(mrun[j], mr);
      alpha[j] = exp2f(mrun[j] - mnew[j]);
      mrun[j]  = mnew[j];
    }
#pragma unroll
    for (int nt = 0; nt < 4; ++nt)
#pragma unroll
      for (int j = 0; j < 8; ++j)
        cf[nt][j] = exp2f(cf[nt][j] - mnew[j]);
#pragma unroll
    for (int j = 0; j < 8; ++j) {
      float rs = rowsum16(cf[0][j] + cf[1][j] + cf[2][j] + cf[3][j]);
      lrun[j] = lrun[j] * alpha[j] + rs;
    }
#pragma unroll
    for (int dt = 0; dt < 8; ++dt)
#pragma unroll
      for (int j = 0; j < 8; ++j)
        oacc[dt][j] *= alpha[j];

    // ---- P: C-layout -> bf16 A-layout via per-wave LDS tile --------------
#pragma unroll
    for (int nt = 0; nt < 4; ++nt)
#pragma unroll
      for (int j = 0; j < 8; ++j)
        sm.Psm[w][j + half8][nt * 16 + l15] = f2bf_u(cf[nt][j]);

    v16bf pf[2];
#pragma unroll
    for (int kc = 0; kc < 2; ++kc) {
      const unsigned short* p0 = &sm.Psm[w][l15][kc * 32 + klo];
      pf[kc] = load_frag16(p0, p0 + 16);
    }

    // ---- O += P V : 8 d-tiles x 2 K-chunks of WMMA bf16 ------------------
#pragma unroll
    for (int dt = 0; dt < 8; ++dt) {
      const unsigned short* vrow = &sm.VTsm[dt * 16 + l15][0];
#pragma unroll
      for (int kc = 0; kc < 2; ++kc) {
        const unsigned short* p0 = vrow + kc * 32 + klo;
        v16bf vb = load_frag16(p0, p0 + 16);
        oacc[dt] = __builtin_amdgcn_wmma_f32_16x16x32_bf16(
            false, pf[kc], false, vb, (short)0, oacc[dt], false, false);
      }
    }
  }

  // ---- finalize: O /= l, write out[b, m, h, d] ----------------------------
#pragma unroll
  for (int j = 0; j < 8; ++j) {
    float inv = 1.0f / lrun[j];
    size_t base = ((size_t)(row0 + j + half8) * NHEADS + h) * D_DIM;
#pragma unroll
    for (int dt = 0; dt < 8; ++dt)
      out[base + dt * 16 + l15] = oacc[dt][j] * inv;
  }
}

// ===========================================================================
extern "C" void kernel_launch(void* const* d_in, const int* in_sizes, int n_in,
                              void* d_out, int out_size, void* d_ws, size_t ws_size,
                              hipStream_t stream) {
  const float* q = (const float*)d_in[0];
  const float* k = (const float*)d_in[1];
  const float* v = (const float*)d_in[2];
  const float* p = (const float*)d_in[3];
  float* out = (float*)d_out;

  // workspace layout (requires ws_size >= 64KB + 2*4MB)
  unsigned int*   msk = (unsigned int*)d_ws;
  unsigned short* kbf = (unsigned short*)((char*)d_ws + 65536);
  unsigned short* vtb = kbf + (size_t)NKV * N_TOT * D_DIM;

  nsa_convert_kernel<<<dim3(N_TOT / 64, NKV), 256, 0, stream>>>(k, v, kbf, vtb);
  nsa_select_kernel<<<(NKV * M_TOT + 255) / 256, 256, 0, stream>>>(p, msk);
  nsa_attn_kernel<<<dim3(M_TOT / 32, NKV), 256, 0, stream>>>(q, kbf, vtb, msk, out);
}
